// final_model_62199716381115
// MI455X (gfx1250) — matrix operand order repaired and checked by reading, not compiled, not run on hardware
//
#include <hip/hip_runtime.h>
#include <hip/hip_bf16.h>
#include <math.h>

typedef _Float16 v16h __attribute__((ext_vector_type(16)));
typedef _Float16 v8h  __attribute__((ext_vector_type(8)));
typedef float    v8f  __attribute__((ext_vector_type(8)));

#define NB 8
#define NN 1024

constexpr int EP_RELU_MASK_F16 = 0;   // C_h = relu(acc + bias[col]) * rowmask[row]
constexpr int EP_PAIRMASK_F16  = 1;   // C_h = acc * rowmask[row] * colmask[col]
constexpr int EP_ACCUM_F32     = 2;   // C_f32 (+)= acc
constexpr int EP_AXPY_F16      = 3;   // C_h = alpha*acc - Sub ; also writes C^T

// ---- CDNA5 async global->LDS copy helpers (ASYNCcnt path) ------------------
__device__ inline unsigned lds_off_u32(const void* p) {
    typedef const __attribute__((address_space(3))) char* as3_t;
    return (unsigned)(unsigned long long)(as3_t)p;   // addrspacecast -> LDS offset
}
__device__ inline void async_b128(unsigned lds, const void* gp) {
    asm volatile("global_load_async_to_lds_b128 %0, %1, off"
                 :: "v"(lds), "v"((unsigned long long)gp)
                 : "memory");
}
__device__ inline void wait_asynccnt0() {
    asm volatile("s_wait_asynccnt 0x0" ::: "memory");
}

// ---------------------------------------------------------------------------
// WMMA GEMM: C[M,Nc] = epilogue(A[M,K] @ B[K,Nc])
//   A:  f16 row-major [M,K].
//   Bt: f16 TRANSPOSED, i.e. [Nc,K] row-major (B[k][n] == Bt[n][k]).
//   Block = 256 threads = 8 waves arranged 4(M) x 2(N).
//   Block tile 128x128, wave tile 32x64, K-step 32: 8 wmma / 12 ds_load_b128.
//   Double-buffered software pipeline: tile s+1 is staged with
//   global_load_async_to_lds_b128 (ASYNCcnt) while tile s is computed; one
//   barrier per K-step. N-edge tiles / K%32 tails use a guarded fallback.
//   Batched via gridDim.z with per-operand strides (0 => shared).
//   Requires M % 128 == 0, K % 16 == 0 (holds for every GEMM in this model).
// ---------------------------------------------------------------------------
template<int EP>
__global__ __launch_bounds__(256) void gemm_wmma(
    const _Float16* __restrict__ Ag, long sA,
    const _Float16* __restrict__ Btg, long sB,
    void* __restrict__ Cg, long sC,
    _Float16* __restrict__ Ctg, long sCt,      // AXPY only: transposed C out
    int M, int K, int Nc,
    const float* __restrict__ bias,
    const float* __restrict__ rowmask, int rmS,
    const float* __restrict__ colmask, int cmS,
    const _Float16* __restrict__ Sub, long sSub,
    float alpha, int accumulate)
{
    __shared__ __align__(16) _Float16 As[2][128][40];   // 128x32 + pad, x2 buffers
    __shared__ __align__(16) _Float16 Bs[2][128][40];   // Bs[.][n][k]

    const int b    = blockIdx.z;
    const _Float16* A  = Ag  + (long)b * sA;
    const _Float16* Bt = Btg + (long)b * sB;
    const int row0 = blockIdx.y * 128;
    const int col0 = blockIdx.x * 128;
    const int t    = threadIdx.x;
    const int lane = t & 31;
    const int wm   = (t >> 5) & 3;   // wave row group: rows wm*32..+31
    const int wn   = t >> 7;         // wave col group: cols wn*64..+63

    v8f acc[2][4] = {};

    const bool interior = (col0 + 128 <= Nc);

    auto stage_async = [&](int k0, int buf) {
        #pragma unroll
        for (int i = 0; i < 2; ++i) {
            const int c  = t + i * 256;          // 0..511
            const int rr = c >> 2;               // 0..127
            const int ks = (c & 3) * 8;          // 0,8,16,24
            async_b128(lds_off_u32(&As[buf][rr][ks]), A  + (long)(row0 + rr) * K + k0 + ks);
            async_b128(lds_off_u32(&Bs[buf][rr][ks]), Bt + (long)(col0 + rr) * K + k0 + ks);
        }
    };
    auto stage_guard = [&](int k0, int buf) {
        #pragma unroll
        for (int i = 0; i < 2; ++i) {
            const int c  = t + i * 256;
            const int rr = c >> 2;
            const int ks = (c & 3) * 8;
            v8h za = {}, zb = {};
            if (k0 + ks < K)
                za = *(const v8h*)(A + (long)(row0 + rr) * K + k0 + ks);
            if ((col0 + rr < Nc) && (k0 + ks < K))
                zb = *(const v8h*)(Bt + (long)(col0 + rr) * K + k0 + ks);
            *(v8h*)&As[buf][rr][ks] = za;
            *(v8h*)&Bs[buf][rr][ks] = zb;
        }
    };
    auto stage = [&](int s, int buf) {
        const int k0 = s * 32;
        if (interior && (k0 + 32 <= K)) stage_async(k0, buf);
        else                            stage_guard(k0, buf);
    };
    auto compute = [&](int buf) {
        const int m = lane & 15, g = lane >> 4;
        v16h af[2];
        #pragma unroll
        for (int i = 0; i < 2; ++i) {
            v8h lo = *(const v8h*)&As[buf][wm * 32 + i * 16 + m][g * 8];
            v8h hi = *(const v8h*)&As[buf][wm * 32 + i * 16 + m][16 + g * 8];
            #pragma unroll
            for (int j = 0; j < 8; ++j) { af[i][j] = lo[j]; af[i][j + 8] = hi[j]; }
        }
        #pragma unroll
        for (int nt = 0; nt < 4; ++nt) {
            const int n = wn * 64 + nt * 16 + m;
            v8h lo = *(const v8h*)&Bs[buf][n][g * 8];
            v8h hi = *(const v8h*)&Bs[buf][n][16 + g * 8];
            v16h bf;
            #pragma unroll
            for (int j = 0; j < 8; ++j) { bf[j] = lo[j]; bf[j + 8] = hi[j]; }
            acc[0][nt] = __builtin_amdgcn_wmma_f32_16x16x32_f16(
                false, af[0], false, bf, (short)0, acc[0][nt], false, false);
            acc[1][nt] = __builtin_amdgcn_wmma_f32_16x16x32_f16(
                false, af[1], false, bf, (short)0, acc[1][nt], false, false);
        }
    };

    // ---- double-buffered pipeline: stage(s+1) overlaps compute(s) ----
    const int nsteps = (K + 31) >> 5;
    stage(0, 0);
    wait_asynccnt0();
    __syncthreads();
    for (int s = 0; s < nsteps; ++s) {
        if (s + 1 < nsteps) stage(s + 1, (s + 1) & 1);
        compute(s & 1);
        wait_asynccnt0();     // next tile landed (no-op on guarded path)
        __syncthreads();      // all waves done reading tile s / writing tile s+1
    }

    // ---- epilogue: VGPR r of lane holds (M = r + 8*(lane>>4), N = lane&15) ----
    const int gg = lane >> 4;
    #pragma unroll
    for (int mi = 0; mi < 2; ++mi) {
        #pragma unroll
        for (int nt = 0; nt < 4; ++nt) {
            const int col = col0 + wn * 64 + nt * 16 + (lane & 15);
            if (col >= Nc) continue;
            #pragma unroll
            for (int rr = 0; rr < 8; ++rr) {
                const int row = row0 + wm * 32 + mi * 16 + rr + 8 * gg;
                float v = acc[mi][nt][rr];
                const long cidx = (long)b * sC + (long)row * Nc + col;
                if (EP == EP_RELU_MASK_F16) {
                    if (bias) v += bias[col];
                    v = v > 0.0f ? v : 0.0f;
                    if (rowmask) v *= rowmask[b * rmS + row];
                    ((_Float16*)Cg)[cidx] = (_Float16)v;
                } else if (EP == EP_PAIRMASK_F16) {
                    v *= rowmask[b * rmS + row] * colmask[b * cmS + col];
                    ((_Float16*)Cg)[cidx] = (_Float16)v;
                } else if (EP == EP_ACCUM_F32) {
                    float* C = (float*)Cg + cidx;
                    *C = (accumulate ? *C : 0.0f) + v;
                } else { // EP_AXPY_F16
                    float s = Sub ? (float)Sub[(long)b * sSub + (long)row * Nc + col] : 0.0f;
                    _Float16 o = (_Float16)(alpha * v - s);
                    ((_Float16*)Cg)[cidx] = o;
                    Ctg[(long)b * sCt + (long)col * M + row] = o;   // transposed twin
                }
            }
        }
    }
}

// ---------------------------------------------------------------------------
// small helper kernels
// ---------------------------------------------------------------------------
__global__ void k_f2h(const float* __restrict__ s, _Float16* __restrict__ d, long n) {
    long i = (long)blockIdx.x * 256 + threadIdx.x;
    if (i < n) d[i] = (_Float16)s[i];
}

// weight transpose+convert: Wt[n][k] = (f16) W[k][n]; z = matrix slice
__global__ void k_w2ht(const float* __restrict__ W, _Float16* __restrict__ Wt, int K, int N) {
    int n = blockIdx.x, z = blockIdx.y;
    const float* Wz = W + (long)z * K * N;
    _Float16* Wtz = Wt + (long)z * K * N;
    for (int k = threadIdx.x; k < K; k += 256)
        Wtz[(long)n * K + k] = (_Float16)Wz[(long)k * N + n];
}

// activation transpose per batch: Xt[b][f][n] = X[b][n][f]
__global__ void k_a2t(const _Float16* __restrict__ X, _Float16* __restrict__ Xt, int F) {
    int f = blockIdx.x, b = blockIdx.y;
    const _Float16* Xb = X + (long)b * NN * F;
    _Float16* Xtb = Xt + (long)b * F * NN;
    for (int n = threadIdx.x; n < NN; n += 256)
        Xtb[(long)f * NN + n] = Xb[(long)n * F + f];
}

// one wave per row: mask = (row sum != 0)
__global__ void k_rowmask(const float* __restrict__ inp, float* __restrict__ mask) {
    int w = threadIdx.x >> 5, lane = threadIdx.x & 31;
    int row = blockIdx.x * 8 + w;
    float s = 0.f;
    const float* p = inp + (long)row * 1024;
    for (int j = lane; j < 1024; j += 32) s += p[j];
    for (int o = 16; o; o >>= 1) s += __shfl_down(s, o, 32);
    if (lane == 0) mask[row] = (s != 0.0f) ? 1.0f : 0.0f;
}

__global__ void k_numel(const float* __restrict__ mask, float* __restrict__ numel) {
    __shared__ float sm[256];
    int b = blockIdx.x, t = threadIdx.x;
    float s = 0.f;
    for (int i = t; i < NN; i += 256) s += mask[b * NN + i];
    sm[t] = s; __syncthreads();
    for (int o = 128; o; o >>= 1) { if (t < o) sm[t] += sm[t + o]; __syncthreads(); }
    if (t == 0) numel[b] = fmaxf(sm[0], 1.0f);
}

__global__ void k_cmean(const _Float16* __restrict__ fea, const float* __restrict__ numel,
                        float* __restrict__ cmean) {
    int b = blockIdx.x;
    for (int f = threadIdx.x; f < 512; f += 256) {
        float s = 0.f;
        const _Float16* p = fea + (long)b * NN * 512 + f;
        for (int n = 0; n < NN; ++n) s += (float)p[(long)n * 512];
        cmean[b * 512 + f] = s / numel[b];
    }
}

__global__ void k_cvec(const float* __restrict__ cmean, const float* __restrict__ Wc,
                       const float* __restrict__ bc, _Float16* __restrict__ c_h) {
    int id = blockIdx.x * 256 + threadIdx.x;
    if (id >= NB * 128) return;
    int b = id >> 7, f = id & 127;
    float s = bc[f];
    for (int k = 0; k < 512; ++k) s += cmean[b * 512 + k] * Wc[k * 128 + f];
    c_h[id] = (_Float16)fmaxf(s, 0.0f);
}

__global__ void k_buildq(const _Float16* __restrict__ fea, const _Float16* __restrict__ c_h,
                         _Float16* __restrict__ q) {
    int row = blockIdx.x, b = row >> 10;
    for (int j = threadIdx.x; j < 640; j += 256)
        q[(long)row * 640 + j] = (j < 512) ? fea[(long)row * 512 + j] : c_h[b * 128 + j - 512];
}

// one wave per row of adjm: dinv = deg>0 ? rsqrt(deg) : 0
__global__ void k_dinv(const _Float16* __restrict__ adjm, float* __restrict__ dinv) {
    int w = threadIdx.x >> 5, lane = threadIdx.x & 31;
    int row = blockIdx.x * 8 + w;
    float s = 0.f;
    const _Float16* p = adjm + (long)row * NN;
    for (int j = lane; j < NN; j += 32) s += (float)p[j];
    for (int o = 16; o; o >>= 1) s += __shfl_down(s, o, 32);
    if (lane == 0) dinv[row] = (s > 0.0f) ? rsqrtf(fmaxf(s, 1e-30f)) : 0.0f;
}

// in place: Lhat = -dinv_i * A * dinv_j
__global__ void k_lhat(_Float16* __restrict__ adjm, const float* __restrict__ dinv) {
    int row = blockIdx.x;
    int base = (row >> 10) << 10;
    float di = dinv[row];
    _Float16* p = adjm + (long)row * NN;
    for (int j = threadIdx.x; j < NN; j += 256)
        p[j] = (_Float16)(-(di * dinv[base + j] * (float)p[j]));
}

__global__ void k_bias_relu(const float* __restrict__ o, const float* __restrict__ bias,
                            _Float16* __restrict__ x, int Nc) {
    int row = blockIdx.x;
    for (int j = threadIdx.x; j < Nc; j += 256)
        x[(long)row * Nc + j] = (_Float16)fmaxf(o[(long)row * Nc + j] + bias[j], 0.0f);
}

// gate[row] = mask ? tanh(g2[row,:64] . Wp3 + bp3) : -1e9   (one wave per row)
__global__ void k_gate(const _Float16* __restrict__ g2, const float* __restrict__ Wp3,
                       const float* __restrict__ bp3, const float* __restrict__ mask,
                       float* __restrict__ gate) {
    int w = threadIdx.x >> 5, lane = threadIdx.x & 31;
    int row = blockIdx.x * 8 + w;
    const _Float16* p = g2 + (long)row * 64;
    float s = (float)p[lane] * Wp3[lane] + (float)p[lane + 32] * Wp3[lane + 32];
    for (int o = 16; o; o >>= 1) s += __shfl_down(s, o, 32);
    if (lane == 0) {
        float t = tanhf(s + bp3[0]);
        gate[row] = (mask[row] > 0.0f) ? t : -1e9f;
    }
}

__global__ void k_softmax(const float* __restrict__ gate, float* __restrict__ attn) {
    __shared__ float sm[256];
    int b = blockIdx.x, t = threadIdx.x;
    float mx = -1e30f;
    for (int i = t; i < NN; i += 256) mx = fmaxf(mx, gate[b * NN + i]);
    sm[t] = mx; __syncthreads();
    for (int o = 128; o; o >>= 1) { if (t < o) sm[t] = fmaxf(sm[t], sm[t + o]); __syncthreads(); }
    mx = sm[0]; __syncthreads();
    float s = 0.f;
    for (int i = t; i < NN; i += 256) s += __expf(gate[b * NN + i] - mx);
    sm[t] = s; __syncthreads();
    for (int o = 128; o; o >>= 1) { if (t < o) sm[t] += sm[t + o]; __syncthreads(); }
    float inv = 1.0f / sm[0];
    for (int i = t; i < NN; i += 256) attn[b * NN + i] = __expf(gate[b * NN + i] - mx) * inv;
}

__global__ void k_pool(const float* __restrict__ attn, const _Float16* __restrict__ x2,
                       float* __restrict__ pooled) {
    int b = blockIdx.x;
    for (int f = threadIdx.x; f < 384; f += 256) {
        float s = 0.f;
        const _Float16* p = x2 + (long)b * NN * 384 + f;
        for (int n = 0; n < NN; ++n) s += attn[b * NN + n] * (float)p[(long)n * 384];
        pooled[b * 384 + f] = s;
    }
}

__global__ void k_head(const float* __restrict__ pooled, const float* __restrict__ Wm1,
                       const float* __restrict__ bm1, const float* __restrict__ Wm2,
                       const float* __restrict__ bm2, float* __restrict__ out) {
    __shared__ float o[256];
    int b = blockIdx.x, t = threadIdx.x;
    float s = bm1[t];
    for (int k = 0; k < 384; ++k) s += pooled[b * 384 + k] * Wm1[k * 256 + t];
    o[t] = fmaxf(s, 0.0f);
    __syncthreads();
    if (t < 32) {
        float a = 0.f;
        for (int j = t; j < 256; j += 32) a += o[j] * Wm2[j];
        for (int off = 16; off; off >>= 1) a += __shfl_down(a, off, 32);
        if (t == 0) out[b] = 1.0f / (1.0f + __expf(-(a + bm2[0])));
    }
}

// ---------------------------------------------------------------------------
extern "C" void kernel_launch(void* const* d_in, const int* in_sizes, int n_in,
                              void* d_out, int out_size, void* d_ws, size_t ws_size,
                              hipStream_t stream) {
    (void)in_sizes; (void)n_in; (void)out_size; (void)ws_size;

    const float* inp = (const float*)d_in[0];
    const float* Wf1 = (const float*)d_in[2];  const float* bf1 = (const float*)d_in[3];
    const float* Wf2 = (const float*)d_in[4];  const float* bf2 = (const float*)d_in[5];
    const float* Wf3 = (const float*)d_in[6];  const float* bf3 = (const float*)d_in[7];
    const float* Wc  = (const float*)d_in[8];  const float* bc  = (const float*)d_in[9];
    const float* Wa1 = (const float*)d_in[10]; const float* ba1 = (const float*)d_in[11];
    const float* Wa2 = (const float*)d_in[12]; const float* ba2 = (const float*)d_in[13];
    const float* Wg1 = (const float*)d_in[14]; const float* bg1 = (const float*)d_in[15];
    const float* Wg2 = (const float*)d_in[16]; const float* bg2 = (const float*)d_in[17];
    const float* Wp1 = (const float*)d_in[18]; const float* bp1 = (const float*)d_in[19];
    const float* Wp2 = (const float*)d_in[20]; const float* bp2 = (const float*)d_in[21];
    const float* Wp3 = (const float*)d_in[22]; const float* bp3 = (const float*)d_in[23];
    const float* Wm1 = (const float*)d_in[24]; const float* bm1 = (const float*)d_in[25];
    const float* Wm2 = (const float*)d_in[26]; const float* bm2 = (const float*)d_in[27];

    char* wp = (char*)d_ws;
    auto alloc = [&](size_t bytes) -> void* {
        void* r = (void*)wp;
        wp += (bytes + 255) & ~(size_t)255;
        return r;
    };
    const long M = (long)NB * NN; // 8192 rows

    _Float16* inp_h = (_Float16*)alloc(M * 1024 * 2);
    _Float16* Wf1t  = (_Float16*)alloc(1024L * 784 * 2);   // all weights stored [N][K] f16
    _Float16* Wf2t  = (_Float16*)alloc(784L * 512 * 2);
    _Float16* Wf3t  = (_Float16*)alloc(512L * 512 * 2);
    _Float16* Wa1t  = (_Float16*)alloc(640L * 512 * 2);
    _Float16* Wa2t  = (_Float16*)alloc(512L * 384 * 2);
    _Float16* Wg1t  = (_Float16*)alloc(5L * 512 * 512 * 2);
    _Float16* Wg2t  = (_Float16*)alloc(5L * 512 * 384 * 2);
    _Float16* Wp1t  = (_Float16*)alloc(384L * 128 * 2);
    _Float16* Wp2t  = (_Float16*)alloc(128L * 64 * 2);
    float* mask  = (float*)alloc(M * 4);
    float* numel = (float*)alloc(NB * 4);
    float* dinv  = (float*)alloc(M * 4);
    _Float16* h1   = (_Float16*)alloc(M * 784 * 2);
    _Float16* h2   = (_Float16*)alloc(M * 512 * 2);
    _Float16* fea  = (_Float16*)alloc(M * 512 * 2);
    float* cmean   = (float*)alloc(NB * 512 * 4);
    _Float16* c_h  = (_Float16*)alloc(NB * 128 * 2);
    _Float16* q    = (_Float16*)alloc(M * 640 * 2);
    _Float16* a1   = (_Float16*)alloc(M * 512 * 2);     // later: Cheb ring (row)
    _Float16* a_h  = (_Float16*)alloc(M * 384 * 2);
    _Float16* adjm = (_Float16*)alloc((long)NB * NN * NN * 2); // becomes Lhat in place
    _Float16* T1   = (_Float16*)alloc(M * 512 * 2);
    _Float16* Xt   = (_Float16*)alloc(M * 512 * 2);     // transposed Cheb ring
    _Float16* T1t  = (_Float16*)alloc(M * 512 * 2);
    _Float16* a1t  = (_Float16*)alloc(M * 512 * 2);
    float* chebout = (float*)alloc(M * 512 * 4);        // reused for layer-2 (384 cols)
    _Float16* x_h  = (_Float16*)alloc(M * 512 * 2);
    _Float16* x2_h = (_Float16*)alloc(M * 384 * 2);
    _Float16* g_h  = (_Float16*)alloc(M * 128 * 2);
    _Float16* g2_h = (_Float16*)alloc(M * 64 * 2);
    float* gate   = (float*)alloc(M * 4);
    float* attn   = (float*)alloc(M * 4);
    float* pooled = (float*)alloc(NB * 384 * 4);

    auto grelu = [&](const _Float16* A, const _Float16* Wt, const float* bias,
                     const float* rmask, _Float16* C, int Mm, int K, int Nc) {
        dim3 grid((Nc + 127) / 128, Mm / 128, 1);
        gemm_wmma<EP_RELU_MASK_F16><<<grid, 256, 0, stream>>>(
            A, 0, Wt, 0, C, 0, nullptr, 0, Mm, K, Nc,
            bias, rmask, 0, nullptr, 0, nullptr, 0, 1.0f, 0);
    };
    auto axpy = [&](const _Float16* Btm, _Float16* Cm, _Float16* Ctm, int Kin,
                    const _Float16* Sub, float alpha) {
        dim3 grid((Kin + 127) / 128, NN / 128, NB);
        gemm_wmma<EP_AXPY_F16><<<grid, 256, 0, stream>>>(
            adjm, (long)NN * NN, Btm, (long)NN * Kin, Cm, (long)NN * Kin,
            Ctm, (long)NN * Kin, NN, NN, Kin,
            nullptr, nullptr, 0, nullptr, 0, Sub, (long)NN * Kin, alpha, 0);
    };
    auto wacc = [&](const _Float16* T, const _Float16* Wkt, int Kin, int Nout, int accum) {
        dim3 grid((Nout + 127) / 128, (unsigned)(M / 128), 1);
        gemm_wmma<EP_ACCUM_F32><<<grid, 256, 0, stream>>>(
            T, 0, Wkt, 0, chebout, 0, nullptr, 0, (int)M, Kin, Nout,
            nullptr, nullptr, 0, nullptr, 0, nullptr, 0, 1.0f, accum);
    };

    // ---- precompute: f16 input, transposed f16 weights, mask ----
    k_f2h<<<(unsigned)((M * 1024 + 255) / 256), 256, 0, stream>>>(inp, inp_h, M * 1024);
    k_w2ht<<<dim3(784, 1), 256, 0, stream>>>(Wf1, Wf1t, 1024, 784);
    k_w2ht<<<dim3(512, 1), 256, 0, stream>>>(Wf2, Wf2t, 784, 512);
    k_w2ht<<<dim3(512, 1), 256, 0, stream>>>(Wf3, Wf3t, 512, 512);
    k_w2ht<<<dim3(512, 1), 256, 0, stream>>>(Wa1, Wa1t, 640, 512);
    k_w2ht<<<dim3(384, 1), 256, 0, stream>>>(Wa2, Wa2t, 512, 384);
    k_w2ht<<<dim3(512, 5), 256, 0, stream>>>(Wg1, Wg1t, 512, 512);
    k_w2ht<<<dim3(384, 5), 256, 0, stream>>>(Wg2, Wg2t, 512, 384);
    k_w2ht<<<dim3(128, 1), 256, 0, stream>>>(Wp1, Wp1t, 384, 128);
    k_w2ht<<<dim3(64, 1), 256, 0, stream>>>(Wp2, Wp2t, 128, 64);
    k_rowmask<<<(unsigned)(M / 8), 256, 0, stream>>>(inp, mask);

    // ---- feature extractor ----
    grelu(inp_h, Wf1t, bf1, mask, h1, (int)M, 1024, 784);
    grelu(h1,    Wf2t, bf2, mask, h2, (int)M, 784, 512);
    grelu(h2,    Wf3t, bf3, mask, fea, (int)M, 512, 512);

    // ---- c vector ----
    k_numel<<<NB, 256, 0, stream>>>(mask, numel);
    k_cmean<<<NB, 256, 0, stream>>>(fea, numel, cmean);
    k_cvec<<<4, 256, 0, stream>>>(cmean, Wc, bc, c_h);

    // ---- adjacency net ----
    k_buildq<<<(unsigned)M, 256, 0, stream>>>(fea, c_h, q);
    grelu(q,  Wa1t, ba1, mask, a1,  (int)M, 640, 512);
    grelu(a1, Wa2t, ba2, mask, a_h, (int)M, 512, 384);

    // ---- adjm = a a^T (a_h is already the [N][K] operand), pair mask folded ----
    {
        dim3 grid(NN / 128, NN / 128, NB);
        gemm_wmma<EP_PAIRMASK_F16><<<grid, 256, 0, stream>>>(
            a_h, (long)NN * 384, a_h, (long)NN * 384, adjm, (long)NN * NN,
            nullptr, 0, NN, 384, NN,
            nullptr, mask, NN, mask, NN, nullptr, 0, 1.0f, 0);
    }
    k_dinv<<<(unsigned)(M / 8), 256, 0, stream>>>(adjm, dinv);
    k_lhat<<<(unsigned)M, 256, 0, stream>>>(adjm, dinv);   // adjm now holds Lhat (f16)

    // ---- Chebyshev layers (K=5) ----
    auto cheb = [&](_Float16* X0, int Kin, int Nout, const _Float16* Wth,
                    const float* bg, _Float16* Xout) {
        k_a2t<<<dim3(Kin, NB), 256, 0, stream>>>(X0, Xt, Kin);   // T0 transposed
        wacc(X0, Wth, Kin, Nout, 0);                             // out  = T0 @ W0
        axpy(Xt, T1, T1t, Kin, nullptr, 1.0f);                   // T1   = Lhat T0
        wacc(T1, Wth + (long)Kin * Nout, Kin, Nout, 1);          // out += T1 @ W1
        _Float16 *Tp = X0, *Tc = T1, *Tct = T1t;
        _Float16* nxt[3]  = { a1,  X0, T1  };
        _Float16* nxtt[3] = { a1t, Xt, T1t };
        for (int k = 2; k < 5; ++k) {
            _Float16* Tn  = nxt[k - 2];
            _Float16* Tnt = nxtt[k - 2];
            axpy(Tct, Tn, Tnt, Kin, Tp, 2.0f);                   // Tk = 2 Lhat T(k-1) - T(k-2)
            wacc(Tn, Wth + (long)k * Kin * Nout, Kin, Nout, 1);
            Tp = Tc; Tc = Tn; Tct = Tnt;
        }
        k_bias_relu<<<(unsigned)M, 256, 0, stream>>>(chebout, bg, Xout, Nout);
    };
    cheb(fea, 512, 512, Wg1t, bg1, x_h);
    cheb(x_h, 512, 384, Wg2t, bg2, x2_h);

    // ---- attention gate + pooling ----
    grelu(x2_h, Wp1t, bp1, nullptr, g_h,  (int)M, 384, 128);
    grelu(g_h,  Wp2t, bp2, nullptr, g2_h, (int)M, 128, 64);
    k_gate<<<(unsigned)(M / 8), 256, 0, stream>>>(g2_h, Wp3, bp3, mask, gate);
    k_softmax<<<NB, 256, 0, stream>>>(gate, attn);
    k_pool<<<NB, 256, 0, stream>>>(attn, x2_h, pooled);

    // ---- output MLP ----
    k_head<<<NB, 256, 0, stream>>>(pooled, Wm1, bm1, Wm2, bm2, (float*)d_out);
}
